// MinGRU_86758339379702
// MI455X (gfx1250) — matrix operand order repaired and checked by reading, not compiled, not run on hardware
//
#include <hip/hip_runtime.h>
#include <hip/hip_bf16.h>

// ---------------------------------------------------------------------------
// minGRU forward, B=8, T=4096, I=1024, H=1024 on gfx1250 (MI455X).
//   k = x@Wz^T + bz ; p = x@Wh^T + bh
//   c_t = sigmoid(-k); v_t = sigmoid(k)*g(p), g(p)=p>=0?p+0.5:sigmoid(p)
//   h_t = c_t*h_{t-1} + v_t, h_{-1} = g(h_0)
// bf16 WMMA dual-GEMM (A tile shared by both weight matrices), gates fused in
// the epilogue, 3-pass chunked linear scan. Double-buffered LDS fed by
// GLOBAL_LOAD_ASYNC_TO_LDS_B128 (no VGPR staging, no spills), 16 v_wmma per
// K-step per wave, one barrier + s_wait_asynccnt per iteration.
// ---------------------------------------------------------------------------

typedef __attribute__((ext_vector_type(16))) __bf16 v16bf;
typedef __attribute__((ext_vector_type(8)))  __bf16 v8bf;
typedef __attribute__((ext_vector_type(8)))  float  v8f;
typedef __attribute__((ext_vector_type(4)))  int    i32x4;

#define AS1 __attribute__((address_space(1)))
#define AS3 __attribute__((address_space(3)))
typedef AS1 i32x4 g_i32x4;   // global-memory 16B vector
typedef AS3 i32x4 l_i32x4;   // LDS 16B vector

#define Bc 8
#define Tc 4096
#define Ic 1024
#define Hc 1024
#define Mc (Bc * Tc)            // 32768 rows
#define M_TILE 128
#define N_TILE 128
#define K_TILE 32
#define LDSS 40                 // padded LDS row stride (elements): 80B = 5*16B
#define TILE_ELEMS (128 * LDSS) // one matrix tile in LDS
#define BUF_ELEMS (3 * TILE_ELEMS)
#define NCHUNK 64
#define LCHUNK (Tc / NCHUNK)    // 64

#if __has_builtin(__builtin_amdgcn_global_load_async_to_lds_b128)
#define HAVE_ASYNC_LDS 1
#else
#define HAVE_ASYNC_LDS 0
#endif

__device__ __forceinline__ void wait_async0() {
#if __has_builtin(__builtin_amdgcn_s_wait_asynccnt)
    __builtin_amdgcn_s_wait_asynccnt(0);
#else
    asm volatile("s_wait_asynccnt 0x0" ::: "memory");
#endif
}

__device__ __forceinline__ unsigned short f2bf(float f) {
    unsigned int u = __float_as_uint(f);
    unsigned int r = u + 0x7FFFu + ((u >> 16) & 1u);   // round-to-nearest-even
    return (unsigned short)(r >> 16);
}

__device__ __forceinline__ float sigmoid_f(float x) {
    return 1.0f / (1.0f + __expf(-x));
}

union V16U { v16bf v; v8bf h[2]; };

// ---------------- conversion: f32 -> bf16 (vectorized x4) ------------------
__global__ __launch_bounds__(256) void cvt_f32_bf16(const float* __restrict__ src,
                                                    unsigned short* __restrict__ dst,
                                                    int n4) {
    int i = blockIdx.x * blockDim.x + threadIdx.x;
    if (i >= n4) return;
    float4 f = reinterpret_cast<const float4*>(src)[i];
    ushort4 o;
    o.x = f2bf(f.x); o.y = f2bf(f.y); o.z = f2bf(f.z); o.w = f2bf(f.w);
    reinterpret_cast<ushort4*>(dst)[i] = o;
}

// --------- fused dual GEMM + gate epilogue (bf16 WMMA, f32 accum) ----------
// grid = (H/N_TILE, M/M_TILE) = (8, 256), block = 256 (8 waves of 32).
// Waves: wm = wave&1 (two 64-row groups), wn = wave>>1 (four 32-col groups).
__global__ __launch_bounds__(256) void gemm_gates(
    const unsigned short* __restrict__ xb,   // [M, I] bf16
    const unsigned short* __restrict__ wzb,  // [H, I] bf16
    const unsigned short* __restrict__ whb,  // [H, I] bf16
    const float* __restrict__ bz,
    const float* __restrict__ bh,
    float* __restrict__ Cws,                 // [M, H] decay coeffs
    float* __restrict__ Vout) {              // [M, H] inputs (in d_out)
    __shared__ __align__(16) unsigned short lds[2 * BUF_ELEMS];  // 60 KiB

    const int n0 = blockIdx.x * N_TILE;
    const int m0 = blockIdx.y * M_TILE;

    const int tid  = threadIdx.x;
    const int lane = tid & 31;
    const int wave = tid >> 5;
    const int wm   = wave & 1;      // 64-row group
    const int wn   = wave >> 1;     // 32-col group
    const int half = lane >> 4;
    const int lr   = lane & 15;

    // tile-load mapping: thread moves 32B per matrix: row = tid>>1, 16-elem half
    const int lrow = tid >> 1;
    const int lcb  = (tid & 1) * 16;

    v8f ak[4][2] = {};  // k accumulators: [row-subtile][col-subtile]
    v8f ah[4][2] = {};  // p accumulators

    const unsigned short* pA = xb  + (size_t)(m0 + lrow) * Ic + lcb;
    const unsigned short* pZ = wzb + (size_t)(n0 + lrow) * Ic + lcb;
    const unsigned short* pH = whb + (size_t)(n0 + lrow) * Ic + lcb;

    // copy one K-step (3 tiles, 32B each per thread) into LDS buffer `dstbuf`
    auto stage = [&](int dstbuf, int ko) {
        unsigned short* db = lds + dstbuf * BUF_ELEMS + lrow * LDSS + lcb;
#if HAVE_ASYNC_LDS
        g_i32x4* ga = (g_i32x4*)(pA + ko);
        g_i32x4* gz = (g_i32x4*)(pZ + ko);
        g_i32x4* gh = (g_i32x4*)(pH + ko);
        l_i32x4* la = (l_i32x4*)db;
        l_i32x4* lz = (l_i32x4*)(db + TILE_ELEMS);
        l_i32x4* lh = (l_i32x4*)(db + 2 * TILE_ELEMS);
        __builtin_amdgcn_global_load_async_to_lds_b128(ga,     la,     0, 0);
        __builtin_amdgcn_global_load_async_to_lds_b128(ga + 1, la + 1, 0, 0);
        __builtin_amdgcn_global_load_async_to_lds_b128(gz,     lz,     0, 0);
        __builtin_amdgcn_global_load_async_to_lds_b128(gz + 1, lz + 1, 0, 0);
        __builtin_amdgcn_global_load_async_to_lds_b128(gh,     lh,     0, 0);
        __builtin_amdgcn_global_load_async_to_lds_b128(gh + 1, lh + 1, 0, 0);
#else
        uint4 a0 = reinterpret_cast<const uint4*>(pA + ko)[0];
        uint4 a1 = reinterpret_cast<const uint4*>(pA + ko)[1];
        uint4 z0 = reinterpret_cast<const uint4*>(pZ + ko)[0];
        uint4 z1 = reinterpret_cast<const uint4*>(pZ + ko)[1];
        uint4 h0 = reinterpret_cast<const uint4*>(pH + ko)[0];
        uint4 h1 = reinterpret_cast<const uint4*>(pH + ko)[1];
        reinterpret_cast<uint4*>(db)[0] = a0;
        reinterpret_cast<uint4*>(db)[1] = a1;
        reinterpret_cast<uint4*>(db + TILE_ELEMS)[0] = z0;
        reinterpret_cast<uint4*>(db + TILE_ELEMS)[1] = z1;
        reinterpret_cast<uint4*>(db + 2 * TILE_ELEMS)[0] = h0;
        reinterpret_cast<uint4*>(db + 2 * TILE_ELEMS)[1] = h1;
#endif
    };

    // prologue: stage K-step 0 into buffer 0
    stage(0, 0);
#if HAVE_ASYNC_LDS
    wait_async0();
#endif
    __syncthreads();

    const int NITER = Ic / K_TILE;  // 32
    int buf = 0;
#pragma unroll 1
    for (int it = 0; it < NITER; ++it) {
        // fetch next K-step into the other buffer (overlaps the WMMA block)
        if (it + 1 < NITER) stage(buf ^ 1, (it + 1) * K_TILE);

        // B fragments stay resident; A fragments streamed per row-subtile
        const unsigned short* bb = lds + buf * BUF_ELEMS;
        V16U bzf[2], bhf[2];
#pragma unroll
        for (int j = 0; j < 2; ++j) {
            const unsigned short* zp = bb + TILE_ELEMS     + (wn * 32 + 16 * j + lr) * LDSS + half * 16;
            const unsigned short* hp = bb + 2 * TILE_ELEMS + (wn * 32 + 16 * j + lr) * LDSS + half * 16;
            bzf[j].h[0] = *reinterpret_cast<const v8bf*>(zp);
            bzf[j].h[1] = *reinterpret_cast<const v8bf*>(zp + 8);
            bhf[j].h[0] = *reinterpret_cast<const v8bf*>(hp);
            bhf[j].h[1] = *reinterpret_cast<const v8bf*>(hp + 8);
        }
#pragma unroll
        for (int i = 0; i < 4; ++i) {
            const unsigned short* ap = bb + (wm * 64 + 16 * i + lr) * LDSS;
            V16U afr;
            afr.h[0] = *reinterpret_cast<const v8bf*>(ap + half * 16);
            afr.h[1] = *reinterpret_cast<const v8bf*>(ap + half * 16 + 8);
            ak[i][0] = __builtin_amdgcn_wmma_f32_16x16x32_bf16(
                false, afr.v, false, bzf[0].v, (short)0, ak[i][0], false, false);
            ah[i][0] = __builtin_amdgcn_wmma_f32_16x16x32_bf16(
                false, afr.v, false, bhf[0].v, (short)0, ah[i][0], false, false);
            ak[i][1] = __builtin_amdgcn_wmma_f32_16x16x32_bf16(
                false, afr.v, false, bzf[1].v, (short)0, ak[i][1], false, false);
            ah[i][1] = __builtin_amdgcn_wmma_f32_16x16x32_bf16(
                false, afr.v, false, bhf[1].v, (short)0, ah[i][1], false, false);
        }

#if HAVE_ASYNC_LDS
        wait_async0();      // our async writes to buf^1 landed in LDS
#endif
        __syncthreads();    // all waves done reading buf / writing buf^1
        buf ^= 1;
    }

    // Epilogue. C/D layout: lane(0-15)=N, VGPR r = row r (lanes 0-15) / r+8 (16-31)
    float bzv[2], bhv[2];
#pragma unroll
    for (int j = 0; j < 2; ++j) {
        const int col = n0 + wn * 32 + 16 * j + lr;
        bzv[j] = bz[col];
        bhv[j] = bh[col];
    }
#pragma unroll
    for (int i = 0; i < 4; ++i) {
        const int rowb = m0 + wm * 64 + 16 * i + half * 8;
#pragma unroll
        for (int j = 0; j < 2; ++j) {
            const int col = n0 + wn * 32 + 16 * j + lr;
#pragma unroll
            for (int r = 0; r < 8; ++r) {
                float kv = ak[i][j][r] + bzv[j];
                float pv = ah[i][j][r] + bhv[j];
                float c  = sigmoid_f(-kv);
                float z  = sigmoid_f(kv);
                float gp = (pv >= 0.0f) ? (pv + 0.5f) : sigmoid_f(pv);
                size_t off = (size_t)(rowb + r) * Hc + col;
                Cws[off]  = c;
                Vout[off] = z * gp;
            }
        }
    }
}

// ----------------- scan pass 1: per-chunk (A, V) reduction -----------------
__global__ __launch_bounds__(256) void scan_chunk(const float* __restrict__ Cws,
                                                  const float* __restrict__ Vin,
                                                  float* __restrict__ chA,
                                                  float* __restrict__ chV) {
    int idx = blockIdx.x * blockDim.x + threadIdx.x;   // B*NCHUNK*H threads
    int h     = idx & (Hc - 1);
    int chunk = (idx >> 10) & (NCHUNK - 1);
    int b     = idx >> 16;
    size_t base = ((size_t)b * Tc + (size_t)chunk * LCHUNK) * Hc + h;
    float A = 1.0f, V = 0.0f;
#pragma unroll 4
    for (int t = 0; t < LCHUNK; ++t) {
        float c = Cws[base + (size_t)t * Hc];
        float v = Vin[base + (size_t)t * Hc];
        A = A * c;
        V = fmaf(c, V, v);
    }
    chA[idx] = A;
    chV[idx] = V;
}

// ------------- scan pass 2: sequential combine across chunks ---------------
__global__ __launch_bounds__(256) void scan_combine(const float* __restrict__ chA,
                                                    const float* __restrict__ chV,
                                                    const float* __restrict__ h0,
                                                    float* __restrict__ hstart) {
    int idx = blockIdx.x * blockDim.x + threadIdx.x;   // B*H threads
    int h = idx & (Hc - 1);
    int b = idx >> 10;
    float x0  = h0[(size_t)b * Hc + h];
    float run = (x0 >= 0.0f) ? (x0 + 0.5f) : sigmoid_f(x0);   // g(h_0)
    size_t base = ((size_t)b * NCHUNK) * Hc + h;
#pragma unroll 4
    for (int c = 0; c < NCHUNK; ++c) {
        size_t o = base + (size_t)c * Hc;
        hstart[o] = run;
        run = fmaf(chA[o], run, chV[o]);
    }
}

// ---------- scan pass 3: replay chunk, write h over v (in d_out) -----------
__global__ __launch_bounds__(256) void scan_apply(const float* __restrict__ Cws,
                                                  const float* __restrict__ hstart,
                                                  float* __restrict__ Vio) {
    int idx = blockIdx.x * blockDim.x + threadIdx.x;   // B*NCHUNK*H threads
    int h     = idx & (Hc - 1);
    int chunk = (idx >> 10) & (NCHUNK - 1);
    int b     = idx >> 16;
    size_t base = ((size_t)b * Tc + (size_t)chunk * LCHUNK) * Hc + h;
    float hp = hstart[idx];
#pragma unroll 4
    for (int t = 0; t < LCHUNK; ++t) {
        size_t o = base + (size_t)t * Hc;
        hp = fmaf(Cws[o], hp, Vio[o]);
        Vio[o] = hp;
    }
}

extern "C" void kernel_launch(void* const* d_in, const int* in_sizes, int n_in,
                              void* d_out, int out_size, void* d_ws, size_t ws_size,
                              hipStream_t stream) {
    const float* x  = (const float*)d_in[0];
    const float* h0 = (const float*)d_in[1];
    const float* Wz = (const float*)d_in[2];
    const float* bz = (const float*)d_in[3];
    const float* Wh = (const float*)d_in[4];
    const float* bh = (const float*)d_in[5];
    float* out = (float*)d_out;

    // workspace layout (bytes)
    char* ws = (char*)d_ws;
    const size_t xb_bytes = (size_t)Mc * Ic * 2;        // 64 MiB bf16 x
    const size_t w_bytes  = (size_t)Hc * Ic * 2;        // 2 MiB each
    const size_t C_bytes  = (size_t)Mc * Hc * 4;        // 128 MiB f32 coeffs
    const size_t ch_elems = (size_t)Bc * NCHUNK * Hc;   // 524288

    unsigned short* xb  = (unsigned short*)ws;
    unsigned short* wzb = (unsigned short*)(ws + xb_bytes);
    unsigned short* whb = (unsigned short*)(ws + xb_bytes + w_bytes);
    float* Cws = (float*)(ws + xb_bytes + 2 * w_bytes);
    float* chA = (float*)(ws + xb_bytes + 2 * w_bytes + C_bytes);
    float* chV = chA + ch_elems;
    float* hst = chV + ch_elems;

    // 1) f32 -> bf16 conversions
    cvt_f32_bf16<<<(Mc * Ic / 4 + 255) / 256, 256, 0, stream>>>(x, xb, Mc * Ic / 4);
    cvt_f32_bf16<<<(Hc * Ic / 4 + 255) / 256, 256, 0, stream>>>(Wz, wzb, Hc * Ic / 4);
    cvt_f32_bf16<<<(Hc * Ic / 4 + 255) / 256, 256, 0, stream>>>(Wh, whb, Hc * Ic / 4);

    // 2) fused dual GEMM + gate epilogue (v -> d_out, c -> ws)
    dim3 ggrid(Hc / N_TILE, Mc / M_TILE);   // (8, 256)
    gemm_gates<<<ggrid, 256, 0, stream>>>(xb, wzb, whb, bz, bh, Cws, out);

    // 3) chunked linear scan
    const int nthr = Bc * NCHUNK * Hc;      // 524288
    scan_chunk<<<nthr / 256, 256, 0, stream>>>(Cws, out, chA, chV);
    scan_combine<<<(Bc * Hc) / 256, 256, 0, stream>>>(chA, chV, h0, hst);
    scan_apply<<<nthr / 256, 256, 0, stream>>>(Cws, hst, out);
}
